// InvariantScalarAttention_1821066133970
// MI455X (gfx1250) — matrix operand order repaired and checked by reading, not compiled, not run on hardware
//
#include <hip/hip_runtime.h>
#include <hip/hip_bf16.h>

typedef __bf16 bf16_t;
typedef __attribute__((ext_vector_type(16))) __bf16 v16bf;
typedef __attribute__((ext_vector_type(8)))  float  v8f;

#define WMMA_BF16(a, b, c) \
  __builtin_amdgcn_wmma_f32_16x16x32_bf16(false, (a), false, (b), (short)0, (c), false, false)

#if defined(__gfx1250__) && __has_builtin(__builtin_amdgcn_global_load_async_to_lds_b128)
#define HAS_ASYNC_LDS 1
typedef int v4i_vs __attribute__((vector_size(4 * sizeof(int))));
typedef __attribute__((address_space(1))) v4i_vs* gptr_v4i;   // global pointer to int4
typedef __attribute__((address_space(3))) v4i_vs* lptr_v4i;   // LDS pointer to int4
#else
#define HAS_ASYNC_LDS 0
#endif

constexpr int kAtoms   = 6144;
constexpr int kDim     = 512;
constexpr int kHeads   = 8;
constexpr int kHeadDim = 64;
constexpr int kGraphs  = 128;

// ---------------- utility kernels ----------------

__global__ void cvt_f32_bf16(const float* __restrict__ src, bf16_t* __restrict__ dst, int count) {
  for (int i = blockIdx.x * blockDim.x + threadIdx.x; i < count; i += gridDim.x * blockDim.x)
    dst[i] = (bf16_t)src[i];
}

__global__ void seg_starts_kernel(const int* __restrict__ batch, int* __restrict__ seg,
                                  int n, int ngraphs) {
  int g = threadIdx.x;
  if (g > ngraphs) return;
  int lo = 0, hi = n;
  while (lo < hi) { int mid = (lo + hi) >> 1; if (batch[mid] < g) lo = mid + 1; else hi = mid; }
  seg[g] = lo;
}

// 16-lane (half-wave) reductions: xor masks < 16 never cross the 16-lane boundary.
static __device__ __forceinline__ float rmax16(float v) {
  v = fmaxf(v, __shfl_xor(v, 1, 32));
  v = fmaxf(v, __shfl_xor(v, 2, 32));
  v = fmaxf(v, __shfl_xor(v, 4, 32));
  v = fmaxf(v, __shfl_xor(v, 8, 32));
  return v;
}
static __device__ __forceinline__ float rsum16(float v) {
  v += __shfl_xor(v, 1, 32);
  v += __shfl_xor(v, 2, 32);
  v += __shfl_xor(v, 4, 32);
  v += __shfl_xor(v, 8, 32);
  return v;
}

// ---------------- tiled WMMA GEMM: C[M,N] = A[M,K](bf16) @ B[K,N](bf16) + bias ----------------
// 128x128 block tile, 256 threads = 8 waves, each wave a 32(M)x64(N) fragment grid (2x4 WMMA tiles).
// A tile staged memory->LDS via gfx1250 async DMA (ASYNCcnt); B tile transposed through VGPRs.

template <bool OUT_BF16>
__launch_bounds__(256, 2)
__global__ void gemm_bias_kernel(const bf16_t* __restrict__ A, const bf16_t* __restrict__ B,
                                 const float* __restrict__ bias, void* __restrict__ Cout,
                                 int M, int N, int K) {
  __shared__ bf16_t As[128][40];   // row stride 80B: 16B-aligned vector stores, conflict-friendly
  __shared__ bf16_t Bt[128][40];   // B tile stored transposed: Bt[n][k]

  const int tid  = threadIdx.x;
  const int lane = tid & 31;
  const int wave = tid >> 5;
  const int wm   = wave & 3;       // 4 waves along M (32 rows each)
  const int wn   = wave >> 2;      // 2 waves along N (64 cols each)
  const int m0   = blockIdx.y * 128;
  const int n0   = blockIdx.x * 128;

  v8f acc[2][4] = {};

  for (int kk = 0; kk < K; kk += 32) {
    __syncthreads();
#pragma unroll
    for (int i = 0; i < 2; ++i) {
      int c = tid + i * 256;                       // 0..511 chunk id (8 bf16 each)
      // A tile: 128 rows x 32 cols, row-major; 16B per lane direct to LDS
      int r = c >> 2, q = (c & 3) << 3;
      unsigned aoff = (unsigned)(m0 + r) * (unsigned)K + (unsigned)(kk + q);
#if HAS_ASYNC_LDS
      __builtin_amdgcn_global_load_async_to_lds_b128(
          (gptr_v4i)(A + aoff), (lptr_v4i)&As[r][q], 0, 0);
#else
      *(uint4*)&As[r][q] = *(const uint4*)(A + aoff);
#endif
      // B tile: read row-major (fixed k, 8 consecutive n), scatter transposed
      int kr = c >> 4, nb = (c & 15) << 3;
      unsigned boff = (unsigned)(kk + kr) * (unsigned)N + (unsigned)(n0 + nb);
      uint4 u = *(const uint4*)(B + boff);
      const bf16_t* pv = (const bf16_t*)&u;
#pragma unroll
      for (int j = 0; j < 8; ++j) Bt[nb + j][kr] = pv[j];
      // prefetch next K-step tiles into cache (global_prefetch_b8)
      if (kk + 32 < K) {
        __builtin_prefetch(A + aoff + 32u, 0, 0);
        __builtin_prefetch(B + boff + 32u * (unsigned)N, 0, 0);
      }
    }
#if HAS_ASYNC_LDS
#if __has_builtin(__builtin_amdgcn_s_wait_asynccnt)
    __builtin_amdgcn_s_wait_asynccnt(0);
#else
    asm volatile("s_wait_asynccnt 0" ::: "memory");
#endif
#endif
    __syncthreads();

    // A fragments: lane holds row (lane&15); lo-half K=0..15, hi-half K=16..31
    v16bf afrag[2];
#pragma unroll
    for (int mt = 0; mt < 2; ++mt) {
      int row = wm * 32 + mt * 16 + (lane & 15);
      int kh = (lane >> 4) * 16;
#pragma unroll
      for (int p = 0; p < 8; ++p) {
        int kp = 2 * p + kh;
        afrag[mt][2 * p]     = As[row][kp];
        afrag[mt][2 * p + 1] = As[row][kp + 1];
      }
    }
    // B fragments: lane holds column (lane&15); lo-half K=0..15, hi-half K=16..31
    v16bf bfrag[4];
#pragma unroll
    for (int nt = 0; nt < 4; ++nt) {
      int col = wn * 64 + nt * 16 + (lane & 15);
      int kb = (lane >> 4) * 16;
#pragma unroll
      for (int e = 0; e < 16; ++e) bfrag[nt][e] = Bt[col][kb + e];
    }
#pragma unroll
    for (int mt = 0; mt < 2; ++mt)
#pragma unroll
      for (int nt = 0; nt < 4; ++nt)
        acc[mt][nt] = WMMA_BF16(afrag[mt], bfrag[nt], acc[mt][nt]);
  }

  // epilogue: C layout -> lane%16 = N, VGPR r + 8*(lane>>4) = M
#pragma unroll
  for (int mt = 0; mt < 2; ++mt)
#pragma unroll
    for (int nt = 0; nt < 4; ++nt) {
      int ncol = n0 + wn * 64 + nt * 16 + (lane & 15);
      float bv = bias[ncol];
      int rbase = m0 + wm * 32 + mt * 16 + 8 * (lane >> 4);
#pragma unroll
      for (int r = 0; r < 8; ++r) {
        float val = acc[mt][nt][r] + bv;
        unsigned idx = (unsigned)(rbase + r) * (unsigned)N + (unsigned)ncol;
        if (OUT_BF16) ((bf16_t*)Cout)[idx] = (bf16_t)val;
        else          ((float*)Cout)[idx]  = val;
      }
    }
}

// ---------------- per-(graph, head) flash attention ----------------
// Block = 4 wave32 waves; each wave owns a 16-query-row tile, loops keys in chunks of 32.
// All global accesses use unsigned 32-bit element offsets (SGPR base + VGPR offset form).

__launch_bounds__(128, 4)
__global__ void attention_kernel(const bf16_t* __restrict__ Q, const bf16_t* __restrict__ Kt,
                                 const bf16_t* __restrict__ V, const int* __restrict__ seg,
                                 bf16_t* __restrict__ ctx) {
  __shared__ bf16_t Plds[4][16][40];   // per-wave P transpose buffer (C layout -> A layout)

  const int g = blockIdx.x;
  const int h = blockIdx.y;
  const int lane = threadIdx.x & 31;
  const int wave = threadIdx.x >> 5;
  const int s0a = seg[g];
  const int nseg = seg[g + 1] - s0a;
  if (nseg <= 0) return;               // uniform: no barriers/WMMA skipped non-uniformly

  const int   hoff  = h * kHeadDim;
  const float scale = 0.125f;          // 1/sqrt(64)
  const int   jc    = lane & 15;
  const int   khalf = lane >> 4;

  for (int i0 = wave * 16; i0 < nseg; i0 += 64) {
    // Q A-fragments for d = 0..31 and 32..63 (row-major Q, contiguous d)
    v16bf qa[2];
    {
      int qrow = i0 + jc;
      unsigned qbase = (unsigned)(s0a + (qrow < nseg ? qrow : nseg - 1)) * kDim + hoff;
      int kh = khalf * 16;
#pragma unroll
      for (int d2 = 0; d2 < 2; ++d2)
#pragma unroll
        for (int p = 0; p < 8; ++p) {
          int kp = 2 * p + kh + d2 * 32;
          qa[d2][2 * p]     = Q[qbase + kp];
          qa[d2][2 * p + 1] = Q[qbase + kp + 1];
        }
    }

    float mrow[8], lrow[8];
    v8f o[4] = {};
#pragma unroll
    for (int r = 0; r < 8; ++r) { mrow[r] = -1e30f; lrow[r] = 0.f; }

    for (int j0 = 0; j0 < nseg; j0 += 32) {
      // S = Q * K^T for a 32-key chunk (two 16-col tiles, K-dim 64 = 2 WMMAs each)
      v8f st[2] = {};
#pragma unroll
      for (int jt = 0; jt < 2; ++jt) {
        int kcol = j0 + jt * 16 + jc;
        unsigned kbase = (unsigned)(s0a + (kcol < nseg ? kcol : nseg - 1)) * kDim + hoff;
        int kb = khalf * 16;
        v16bf b0, b1;
#pragma unroll
        for (int e = 0; e < 16; ++e) {
          b0[e] = Kt[kbase + kb + e];
          b1[e] = Kt[kbase + 32 + kb + e];
        }
        st[jt] = WMMA_BF16(qa[0], b0, st[jt]);
        st[jt] = WMMA_BF16(qa[1], b1, st[jt]);
      }

      // online softmax (per-row state; row = VGPR r within a 16-lane half)
      bool v0 = (j0 + jc) < nseg;
      bool v1 = (j0 + 16 + jc) < nseg;
#pragma unroll
      for (int r = 0; r < 8; ++r) {
        float a0 = v0 ? st[0][r] * scale : -1e30f;
        float a1 = v1 ? st[1][r] * scale : -1e30f;
        float mx = rmax16(fmaxf(a0, a1));
        float mn = fmaxf(mrow[r], mx);
        float p0 = __expf(a0 - mn);
        float p1 = __expf(a1 - mn);
        float corr = __expf(mrow[r] - mn);
        lrow[r] = lrow[r] * corr + rsum16(p0 + p1);
        mrow[r] = mn;
#pragma unroll
        for (int dt = 0; dt < 4; ++dt) o[dt][r] *= corr;
        int prow = r + 8 * khalf;
        Plds[wave][prow][jc]      = (bf16_t)p0;
        Plds[wave][prow][16 + jc] = (bf16_t)p1;
      }
      asm volatile("s_wait_dscnt 0" ::: "memory");  // cross-lane LDS write->read inside one wave

      // re-read P in A-fragment layout
      v16bf pa;
      {
        int kh = khalf * 16;
#pragma unroll
        for (int p = 0; p < 8; ++p) {
          int kp = 2 * p + kh;
          pa[2 * p]     = Plds[wave][jc][kp];
          pa[2 * p + 1] = Plds[wave][jc][kp + 1];
        }
      }
      // O += P(16x32) * V(32x16) per 16-wide d tile; masked keys have P==0
      int kb = khalf * 16;
#pragma unroll
      for (int dt = 0; dt < 4; ++dt) {
        v16bf vb;
        int dcol = hoff + dt * 16 + jc;
#pragma unroll
        for (int e = 0; e < 16; ++e) {
          int kk2 = j0 + kb + e;
          unsigned voff = (unsigned)(s0a + (kk2 < nseg ? kk2 : nseg - 1)) * kDim + dcol;
          vb[e] = V[voff];
        }
        o[dt] = WMMA_BF16(pa, vb, o[dt]);
      }
      asm volatile("s_wait_dscnt 0" ::: "memory");  // reads retired before next-iter P writes
    }

    // normalize + store context (bf16) for the output projection
#pragma unroll
    for (int dt = 0; dt < 4; ++dt) {
      int dcol = hoff + dt * 16 + jc;
#pragma unroll
      for (int r = 0; r < 8; ++r) {
        int row = i0 + r + 8 * khalf;
        if (row < nseg)
          ctx[(unsigned)(s0a + row) * kDim + dcol] = (bf16_t)(o[dt][r] / lrow[r]);
      }
    }
  }
}

// ---------------- host side ----------------

extern "C" void kernel_launch(void* const* d_in, const int* in_sizes, int n_in,
                              void* d_out, int out_size, void* d_ws, size_t ws_size,
                              hipStream_t stream) {
  const float* h     = (const float*)d_in[0];
  const int*   batch = (const int*)d_in[1];
  const float* Wq    = (const float*)d_in[2];
  const float* bq    = (const float*)d_in[3];
  const float* Wk    = (const float*)d_in[4];
  const float* bk    = (const float*)d_in[5];
  const float* Wv    = (const float*)d_in[6];
  const float* bv    = (const float*)d_in[7];
  const float* Wo    = (const float*)d_in[8];
  const float* bo    = (const float*)d_in[9];

  char* ws = (char*)d_ws;
  size_t off = 0;
  auto take = [&](size_t bytes) -> void* {
    void* p = ws + off;
    off += (bytes + 255) & ~(size_t)255;
    return p;
  };

  const size_t HB = (size_t)kAtoms * kDim;   // elements
  const size_t WB = (size_t)kDim * kDim;

  bf16_t* hbf = (bf16_t*)take(HB * 2);
  bf16_t* Wqb = (bf16_t*)take(WB * 2);
  bf16_t* Wkb = (bf16_t*)take(WB * 2);
  bf16_t* Wvb = (bf16_t*)take(WB * 2);
  bf16_t* Wob = (bf16_t*)take(WB * 2);
  bf16_t* Qb  = (bf16_t*)take(HB * 2);
  bf16_t* Kb  = (bf16_t*)take(HB * 2);
  bf16_t* Vb  = (bf16_t*)take(HB * 2);
  bf16_t* Cb  = (bf16_t*)take(HB * 2);
  int*    seg = (int*)take((kGraphs + 1) * sizeof(int));

  cvt_f32_bf16<<<1024, 256, 0, stream>>>(h,  hbf, (int)HB);
  cvt_f32_bf16<<<256, 256, 0, stream>>>(Wq, Wqb, (int)WB);
  cvt_f32_bf16<<<256, 256, 0, stream>>>(Wk, Wkb, (int)WB);
  cvt_f32_bf16<<<256, 256, 0, stream>>>(Wv, Wvb, (int)WB);
  cvt_f32_bf16<<<256, 256, 0, stream>>>(Wo, Wob, (int)WB);
  seg_starts_kernel<<<1, kGraphs + 1, 0, stream>>>(batch, seg, kAtoms, kGraphs);

  dim3 gg(kDim / 128, kAtoms / 128);  // (4, 48)
  gemm_bias_kernel<true><<<gg, 256, 0, stream>>>(hbf, Wqb, bq, Qb, kAtoms, kDim, kDim);
  gemm_bias_kernel<true><<<gg, 256, 0, stream>>>(hbf, Wkb, bk, Kb, kAtoms, kDim, kDim);
  gemm_bias_kernel<true><<<gg, 256, 0, stream>>>(hbf, Wvb, bv, Vb, kAtoms, kDim, kDim);

  attention_kernel<<<dim3(kGraphs, kHeads), 128, 0, stream>>>(Qb, Kb, Vb, seg, Cb);

  gemm_bias_kernel<false><<<gg, 256, 0, stream>>>(Cb, Wob, bo, d_out, kAtoms, kDim, kDim);

  (void)in_sizes; (void)n_in; (void)out_size; (void)ws_size;
}
